// GestureNet_42417097017265
// MI455X (gfx1250) — matrix operand order repaired and checked by reading, not compiled, not run on hardware
//
#include <hip/hip_runtime.h>
#include <hip/hip_bf16.h>
#include <math.h>

typedef __attribute__((ext_vector_type(16))) __bf16 v16bf;
typedef __attribute__((ext_vector_type(8)))  float  v8f;
typedef __attribute__((ext_vector_type(4)))  unsigned int v4ui;
typedef __attribute__((ext_vector_type(8)))  int v8i_t;
typedef __attribute__((ext_vector_type(4)))  int v4i_t;

// ---------------- model dims ----------------
#define BATCH 128
#define TFULL 2048
#define FIN   8
#define CF1   128
#define T1    1024       // after pool1
#define CF2   256
#define T2    512        // after pool2
#define HID   96
#define G4    384        // 4*HID
#define DL    256        // lstm input dim
#define ROWS  132        // conv2 strip tile rows: 128 + 2 halo each side

// ---------------- ws layout (bytes) ----------------
#define OFF_ACT1 ((size_t)0)                 // bf16 [128][1024][128]   33,554,432
#define OFF_ACT2 ((size_t)100663296)         // bf16 [128][512][256]    33,554,432
#define OFF_XP   ((size_t)134217728)         // f32  [2][512][128][384] 201,326,592
#define OFF_LO   ((size_t)335544320)         // f32  [128][512][192]    50,331,648
#define OFF_W2A  ((size_t)385875968)         // bf16 packed conv2 A        327,680
#define OFF_WIA  ((size_t)386203648)         // bf16 packed wih A (2 dir)  393,216
#define OFF_WHA  ((size_t)386596864)         // bf16 packed whh A (2 dir)  147,456

#define N_C2A (20*16*32*16)
#define N_WIA (8*24*32*16)
#define N_WHA (3*24*32*16)

// ---------------- helpers ----------------
__device__ __forceinline__ float sigm(float x) { return 1.0f / (1.0f + expf(-x)); }

__device__ __forceinline__ float waveRedSum(float v) {
  #pragma unroll
  for (int o = 16; o > 0; o >>= 1) v += __shfl_xor(v, o, 32);
  return v;
}

__device__ __forceinline__ v8f wmma_bf16(v16bf a, v16bf b, v8f c) {
  return __builtin_amdgcn_wmma_f32_16x16x32_bf16(false, a, false, b, (short)0, c, false, false);
}

// Lane layout of 16-bit 16x32 A fragments (matches contiguous B loads):
// lane = hf*16 + row; element e
// K = hf*16 + e
__device__ __forceinline__ int a_klocal(int hf, int e) {
  return hf * 16 + e;
}

// ---- Tensor Data Mover: 2-D tile (rows x 128 bf16) global -> LDS ----
// D# per CDNA5 ISA ch.8: group0 {count=1, lds_addr, global_addr, type=2},
// group1 {data_size=2B, tensor_dim0/1, tile_dim0/1, dim0 stride}. OOB rows
// beyond tensor_dim1 are zero-filled by the TDM.
__device__ __forceinline__ void tdm_tile_load(unsigned lds_off, __bf16* lds_ptr,
                                              const __bf16* gsrc,
                                              unsigned rows_req, unsigned rows_avail,
                                              int lane) {
#if __has_builtin(__builtin_amdgcn_tensor_load_to_lds)
  unsigned long long gaddr = (unsigned long long)gsrc;
  v4ui g0;
  g0[0] = 1u;                                            // count=1 (user mode)
  g0[1] = lds_off;                                       // lds_addr (bytes)
  g0[2] = (unsigned)(gaddr & 0xFFFFFFFFu);               // global_addr[31:0]
  g0[3] = (unsigned)((gaddr >> 32) & 0x1FFFFFFu) | (2u << 30); // addr[56:32] | type=2
  v8i_t g1;
  g1[0] = (int)(1u << 16);                               // data_size=1 (2 bytes)
  g1[1] = (int)((CF1 & 0xFFFFu) << 16);                  // tensor_dim0[15:0]
  g1[2] = (int)(((unsigned)CF1 >> 16) | ((rows_avail & 0xFFFFu) << 16)); // dim0 hi | dim1 lo
  g1[3] = (int)(((rows_avail >> 16) & 0xFFFFu) | ((unsigned)CF1 << 16)); // dim1 hi | tile_dim0
  g1[4] = (int)(rows_req & 0xFFFFu);                     // tile_dim1 (tile_dim2=0)
  g1[5] = (int)(unsigned)CF1;                            // tensor_dim0_stride[31:0]
  g1[6] = 0;                                             // stride hi / dim1_stride lo
  g1[7] = 0;
  v4i_t g2 = {0, 0, 0, 0};
  v4i_t g3 = {0, 0, 0, 0};
#if defined(__clang_major__) && (__clang_major__ >= 23)
  v8i_t g4 = {0, 0, 0, 0, 0, 0, 0, 0};
  __builtin_amdgcn_tensor_load_to_lds(g0, g1, g2, g3, g4, 0);
#else
  __builtin_amdgcn_tensor_load_to_lds(g0, g1, g2, g3, 0);
#endif
  (void)lds_ptr; (void)lane;
#else
  // fallback: cooperative copy by the issuing wave (32 lanes)
  for (unsigned i = (unsigned)lane; i < rows_req * CF1; i += 32) {
    unsigned r = i / CF1;
    lds_ptr[i] = (r < rows_avail) ? gsrc[i] : (__bf16)0.0f;
  }
#endif
}

__device__ __forceinline__ void tdm_wait0() {
#if __has_builtin(__builtin_amdgcn_s_wait_tensorcnt)
  __builtin_amdgcn_s_wait_tensorcnt(0);
#endif
}

// ---------------- weight packing ----------------
__global__ void pack_conv2A(const float* __restrict__ w, __bf16* __restrict__ dst) {
  int idx = blockIdx.x * blockDim.x + threadIdx.x;
  if (idx >= N_C2A) return;
  int e = idx & 15, l = (idx >> 4) & 31, mt = (idx >> 9) & 15, c = idx >> 13; // c in [0,20)
  int co = mt * 16 + (l & 15);
  int kl = a_klocal(l >> 4, e);
  int tap = c >> 2;
  int ci = (c & 3) * 32 + kl;
  dst[idx] = (__bf16)w[(co * CF1 + ci) * 5 + tap];
}

__global__ void pack_wihA(const float* __restrict__ w, __bf16* __restrict__ dst) {
  int idx = blockIdx.x * blockDim.x + threadIdx.x;
  if (idx >= N_WIA) return;
  int e = idx & 15, l = (idx >> 4) & 31;
  int rest = idx >> 9;
  int mt = rest % 24;
  int c = rest / 24;          // [0,8)
  int m = mt * 16 + (l & 15);
  int K = c * 32 + a_klocal(l >> 4, e);
  dst[idx] = (__bf16)w[m * DL + K];
}

__global__ void pack_whhA(const float* __restrict__ w, __bf16* __restrict__ dst) {
  int idx = blockIdx.x * blockDim.x + threadIdx.x;
  if (idx >= N_WHA) return;
  int e = idx & 15, l = (idx >> 4) & 31;
  int rest = idx >> 9;
  int mt = rest % 24;
  int c = rest / 24;          // [0,3)
  int m = mt * 16 + (l & 15);
  int K = c * 32 + a_klocal(l >> 4, e);
  dst[idx] = (__bf16)w[m * HID + K];
}

// ---------------- conv1 + BN + ReLU + pool (small, VALU) ----------------
__global__ void conv1_bn_pool(const float* __restrict__ x, const float* __restrict__ w,
                              const float* __restrict__ cb,
                              const float* __restrict__ g1, const float* __restrict__ b1,
                              const float* __restrict__ m1, const float* __restrict__ v1,
                              __bf16* __restrict__ out) {
  int id = blockIdx.x * blockDim.x + threadIdx.x;
  if (id >= BATCH * T1 * CF1) return;
  int co = id & 127;
  int tp = (id >> 7) & 1023;
  int b  = id >> 17;
  float res[2];
  #pragma unroll
  for (int pp = 0; pp < 2; ++pp) {
    int t0 = 2 * tp + pp;
    float acc = cb[co];
    for (int k = 0; k < 7; ++k) {
      int ts = t0 + k - 3;
      if ((unsigned)ts < (unsigned)TFULL) {
        const float* xr = x + ((size_t)b * TFULL + ts) * FIN;
        const float* wr = w + (co * FIN) * 7 + k;
        #pragma unroll
        for (int ci = 0; ci < 8; ++ci) acc += xr[ci] * wr[ci * 7];
      }
    }
    res[pp] = acc;
  }
  float sc = g1[co] * rsqrtf(v1[co] + 1e-5f);
  float y0 = fmaxf((res[0] - m1[co]) * sc + b1[co], 0.f);
  float y1 = fmaxf((res[1] - m1[co]) * sc + b1[co], 0.f);
  out[((size_t)b * T1 + tp) * CF1 + co] = (__bf16)fmaxf(y0, y1);
}

// ---------------- conv2 implicit GEMM: one TDM strip tile + WMMA + fused pool ----
// Workgroup (8 waves) computes 256 out-ch x 128 positions. The 132x128 bf16
// activation strip (33KB) is DMA'd once by the TDM. K-chunk is the OUTER loop:
// each A-fragment pair is loaded once and reused for 8 t-tiles (16 WMMAs),
// 16 v8f accumulators stay in registers. Maxpool fused via __shfl_xor.
__global__ void __launch_bounds__(256, 1)
conv2_wmma_tdm(const __bf16* __restrict__ act1, const __bf16* __restrict__ wA,
               const float* __restrict__ cb,
               const float* __restrict__ bn_g, const float* __restrict__ bn_b,
               const float* __restrict__ bn_m, const float* __restrict__ bn_v,
               __bf16* __restrict__ act2out) {
  __shared__ __bf16 tilebuf[ROWS * CF1];    // 33,792 B
  int b     = blockIdx.x >> 3;
  int strip = blockIdx.x & 7;
  int tid = threadIdx.x;
  int lane = tid & 31, wave = tid >> 5;
  int hf = lane >> 4, ln = lane & 15;
  int tbase = strip * 128;

  if (wave == 0) {
    int row_start = tbase - 2;
    int clampv = row_start < 0 ? 0 : row_start;
    int skip = clampv - row_start;                   // 0 or 2 (head pad)
    unsigned rows_req = (unsigned)(ROWS - skip);
    unsigned rows_avail = (unsigned)(T1 - clampv);   // tail OOB -> TDM zero-fill
    for (int z = lane; z < skip * CF1; z += 32) tilebuf[z] = (__bf16)0.0f;
    unsigned lds_off = ((unsigned)(unsigned long long)(void*)&tilebuf[0]) +
                       (unsigned)(skip * CF1 * 2);
    tdm_tile_load(lds_off, &tilebuf[(size_t)skip * CF1],
                  act1 + ((size_t)b * T1 + clampv) * CF1, rows_req, rows_avail, lane);
    tdm_wait0();
  }
  __syncthreads();

  int mt0 = wave * 2, mt1 = wave * 2 + 1;
  v8f zv;
  #pragma unroll
  for (int k = 0; k < 8; ++k) zv[k] = 0.f;
  v8f acc[8][2];
  #pragma unroll
  for (int i = 0; i < 8; ++i) { acc[i][0] = zv; acc[i][1] = zv; }

  #pragma unroll 1
  for (int c = 0; c < 20; ++c) {
    int tap = c >> 2;
    int ci0 = (c & 3) * 32 + hf * 16;
    v16bf a0 = *(const v16bf*)(wA + ((size_t)(c * 16 + mt0) * 32 + lane) * 16);
    v16bf a1 = *(const v16bf*)(wA + ((size_t)(c * 16 + mt1) * 32 + lane) * 16);
    const __bf16* bbase = tilebuf + (ln + tap) * CF1 + ci0;
    #pragma unroll
    for (int i = 0; i < 8; ++i) {
      v16bf bfrag = *(const v16bf*)(bbase + (size_t)i * 16 * CF1);
      acc[i][0] = wmma_bf16(a0, bfrag, acc[i][0]);
      acc[i][1] = wmma_bf16(a1, bfrag, acc[i][1]);
    }
  }

  // epilogue: bias + BN + ReLU, fused 2:1 maxpool across adjacent t lanes
  #pragma unroll
  for (int mm = 0; mm < 2; ++mm) {
    int mt = wave * 2 + mm;
    #pragma unroll
    for (int r = 0; r < 8; ++r) {
      int co = mt * 16 + hf * 8 + r;
      float sc = bn_g[co] * rsqrtf(bn_v[co] + 1e-5f);
      float shift = (cb[co] - bn_m[co]) * sc + bn_b[co];
      #pragma unroll
      for (int i = 0; i < 8; ++i) {
        float val = fmaxf(acc[i][mm][r] * sc + shift, 0.f);
        float vmax = fmaxf(val, __shfl_xor(val, 1, 32));
        if ((ln & 1) == 0) {
          int tp = (tbase + i * 16 + ln) >> 1;
          act2out[((size_t)b * T2 + tp) * CF2 + co] = (__bf16)vmax;
        }
      }
    }
  }
}

// ---------------- LSTM input projection: xp = seq @ Wih^T + bih (WMMA) ----------------
__global__ void xp_gemm(const __bf16* __restrict__ a2, const __bf16* __restrict__ wA,
                        const float* __restrict__ fbih, const float* __restrict__ bbih,
                        float* __restrict__ xp) {
  int wid = (blockIdx.x * blockDim.x + threadIdx.x) >> 5;
  int lane = threadIdx.x & 31;
  int nt = wid & 7;
  int rest = wid >> 3;
  int mt = rest % 24; rest /= 24;
  int t = rest & 511;
  int dir = rest >> 9;
  int hf = lane >> 4, ln = lane & 15;
  int b = nt * 16 + ln;
  const __bf16* bp = a2 + ((size_t)b * T2 + t) * DL + hf * 16;
  v8f acc = {};
  #pragma unroll
  for (int c = 0; c < 8; ++c) {
    v16bf afrag = *(const v16bf*)(wA + (((size_t)(dir * 8 + c) * 24 + mt) * 32 + lane) * 16);
    v16bf bfrag = *(const v16bf*)(bp + c * 32);
    acc = wmma_bf16(afrag, bfrag, acc);
  }
  const float* bih = dir ? bbih : fbih;
  #pragma unroll
  for (int r = 0; r < 8; ++r) {
    int m = mt * 16 + hf * 8 + r;
    xp[(((size_t)dir * T2 + t) * BATCH + b) * G4 + m] = acc[r] + bih[m];
  }
}

// ---------------- LN over 384 (one wave per row) ----------------
__global__ void xp_ln(float* __restrict__ xp,
                      const float* __restrict__ fg, const float* __restrict__ fb,
                      const float* __restrict__ bg, const float* __restrict__ bb) {
  int row = (blockIdx.x * blockDim.x + threadIdx.x) >> 5; // [0, 131072)
  int lane = threadIdx.x & 31;
  int dir = row >> 16;
  float* p = xp + (size_t)row * G4;
  const float* g = dir ? bg : fg;
  const float* b = dir ? bb : fb;
  float vals[12], s = 0.f, s2 = 0.f;
  #pragma unroll
  for (int i = 0; i < 12; ++i) {
    float v = p[lane + 32 * i];
    vals[i] = v; s += v; s2 += v * v;
  }
  s = waveRedSum(s); s2 = waveRedSum(s2);
  float mean = s * (1.f / 384.f);
  float var = s2 * (1.f / 384.f) - mean * mean;
  float inv = rsqrtf(var + 1e-5f);
  #pragma unroll
  for (int i = 0; i < 12; ++i) {
    int j = lane + 32 * i;
    p[j] = (vals[i] - mean) * inv * g[j] + b[j];
  }
}

// ---------------- persistent layernorm-LSTM scan (1 WG per direction) ----------------
__global__ void __launch_bounds__(1024, 1)
lstm_scan(const float* __restrict__ xp, const __bf16* __restrict__ whhA,
          const float* __restrict__ fbhh, const float* __restrict__ flg, const float* __restrict__ flb,
          const float* __restrict__ fog, const float* __restrict__ fob,
          const float* __restrict__ bbhh, const float* __restrict__ blg, const float* __restrict__ blb,
          const float* __restrict__ bog, const float* __restrict__ bob,
          float* __restrict__ lstm_out) {
  extern __shared__ char smem[];
  float*  gates = (float*)smem;                                   // 128*384 f32
  float*  cst   = (float*)(smem + (size_t)BATCH * G4 * 4);        // 128*96 f32
  __bf16* hst   = (__bf16*)(smem + (size_t)BATCH * G4 * 4 + (size_t)BATCH * HID * 4);

  int dir = blockIdx.x;
  const float* bhh = dir ? bbhh : fbhh;
  const float* lg  = dir ? blg  : flg;
  const float* lb  = dir ? blb  : flb;
  const float* og  = dir ? bog  : fog;
  const float* ob  = dir ? bob  : fob;

  int tid = threadIdx.x;
  int lane = tid & 31, wave = tid >> 5;
  int hf = lane >> 4, ln = lane & 15;

  for (int i = tid; i < BATCH * HID; i += 1024) { cst[i] = 0.f; hst[i] = (__bf16)0.f; }
  __syncthreads();

  for (int step = 0; step < T2; ++step) {
    int t = dir ? (T2 - 1 - step) : step;

    // ---- phase 1: gates_raw = h @ Whh^T  (192 tiles, 6/wave, 3 WMMA each) ----
    #pragma unroll 2
    for (int it = 0; it < 6; ++it) {
      int tile = wave * 6 + it;
      int mt = tile % 24, nt = tile / 24;
      const __bf16* bp = hst + (nt * 16 + ln) * HID + hf * 16;
      v8f acc = {};
      #pragma unroll
      for (int c = 0; c < 3; ++c) {
        v16bf afrag = *(const v16bf*)(whhA + (((size_t)(dir * 3 + c) * 24 + mt) * 32 + lane) * 16);
        v16bf bfrag = *(const v16bf*)(bp + c * 32);
        acc = wmma_bf16(afrag, bfrag, acc);
      }
      int bcol = nt * 16 + ln;
      #pragma unroll
      for (int r = 0; r < 8; ++r) {
        int m = mt * 16 + hf * 8 + r;
        gates[bcol * G4 + m] = acc[r];
      }
    }
    __syncthreads();

    // ---- phase 2: per-row LN(384) + gates + cell + LN(96) + h ----
    for (int rr = 0; rr < 4; ++rr) {
      int b = wave * 4 + rr;
      float* grow = gates + b * G4;
      const float* xrow = xp + (((size_t)dir * T2 + t) * BATCH + b) * G4;

      float vals[12], s = 0.f, s2 = 0.f;
      #pragma unroll
      for (int i = 0; i < 12; ++i) {
        int j = lane + 32 * i;
        float v = grow[j] + bhh[j];
        vals[i] = v; s += v; s2 += v * v;
      }
      s = waveRedSum(s); s2 = waveRedSum(s2);
      float mean = s * (1.f / 384.f);
      float var = s2 * (1.f / 384.f) - mean * mean;
      float inv = rsqrtf(var + 1e-5f);
      float gv[12];
      #pragma unroll
      for (int i = 0; i < 12; ++i) {
        int j = lane + 32 * i;
        gv[i] = (vals[i] - mean) * inv * lg[j] + lb[j] + xrow[j];
      }
      // hidden unit j = lane + 32*s3 -> gates i/f/g/o at slots s3, s3+3, s3+6, s3+9
      float cn[3], cs = 0.f, cs2 = 0.f;
      #pragma unroll
      for (int s3 = 0; s3 < 3; ++s3) {
        int j = lane + 32 * s3;
        float ig = sigm(gv[s3]);
        float fg2 = sigm(gv[s3 + 3]);
        float gg = tanhf(gv[s3 + 6]);
        float cc = fg2 * cst[b * HID + j] + ig * gg;
        cn[s3] = cc; cs += cc; cs2 += cc * cc;
      }
      cs = waveRedSum(cs); cs2 = waveRedSum(cs2);
      float cmean = cs * (1.f / 96.f);
      float cvar = cs2 * (1.f / 96.f) - cmean * cmean;
      float cinv = rsqrtf(cvar + 1e-5f);
      #pragma unroll
      for (int s3 = 0; s3 < 3; ++s3) {
        int j = lane + 32 * s3;
        float lnc = (cn[s3] - cmean) * cinv * og[j] + ob[j];
        float hv = sigm(gv[s3 + 9]) * tanhf(lnc);
        cst[b * HID + j] = cn[s3];
        hst[b * HID + j] = (__bf16)hv;
        lstm_out[((size_t)b * T2 + t) * (2 * HID) + dir * HID + j] = hv;
      }
    }
    __syncthreads();
  }
}

// ---------------- attention + fc1(BN,ReLU) + fc2 ----------------
__global__ void head(const float* __restrict__ lo, const float* __restrict__ attn_w,
                     const float* __restrict__ fc1w, const float* __restrict__ fc1b,
                     const float* __restrict__ g3, const float* __restrict__ b3,
                     const float* __restrict__ m3, const float* __restrict__ v3,
                     const float* __restrict__ fc2w, const float* __restrict__ fc2b,
                     float* __restrict__ out) {
  __shared__ float logit[512];
  __shared__ float red[256];
  __shared__ float ctx[192];
  __shared__ float h2[256];
  int b = blockIdx.x, tid = threadIdx.x;
  const float* base = lo + (size_t)b * T2 * (2 * HID);
  for (int t = tid; t < 512; t += 256) {
    float s = 0.f;
    const float* row = base + t * 192;
    for (int j = 0; j < 192; ++j) s += row[j] * attn_w[j];
    logit[t] = s;
  }
  __syncthreads();
  float mx = fmaxf(logit[tid], logit[tid + 256]);
  red[tid] = mx; __syncthreads();
  for (int o = 128; o > 0; o >>= 1) { if (tid < o) red[tid] = fmaxf(red[tid], red[tid + o]); __syncthreads(); }
  mx = red[0]; __syncthreads();
  float e0 = expf(logit[tid] - mx), e1 = expf(logit[tid + 256] - mx);
  logit[tid] = e0; logit[tid + 256] = e1;
  red[tid] = e0 + e1; __syncthreads();
  for (int o = 128; o > 0; o >>= 1) { if (tid < o) red[tid] += red[tid + o]; __syncthreads(); }
  float inv = 1.f / red[0]; __syncthreads();
  if (tid < 192) {
    float s = 0.f;
    for (int t = 0; t < 512; ++t) s += logit[t] * base[t * 192 + tid];
    ctx[tid] = s * inv;
  }
  __syncthreads();
  {
    float s = fc1b[tid];
    for (int j = 0; j < 192; ++j) s += ctx[j] * fc1w[tid * 192 + j];
    float sc = g3[tid] * rsqrtf(v3[tid] + 1e-5f);
    h2[tid] = fmaxf((s - m3[tid]) * sc + b3[tid], 0.f);
  }
  __syncthreads();
  if (tid < 6) {
    float s = fc2b[tid];
    for (int j = 0; j < 256; ++j) s += h2[j] * fc2w[tid * 256 + j];
    out[b * 6 + tid] = s;
  }
}

// ---------------- launcher ----------------
extern "C" void kernel_launch(void* const* d_in, const int* in_sizes, int n_in,
                              void* d_out, int out_size, void* d_ws, size_t ws_size,
                              hipStream_t stream) {
  const float* x       = (const float*)d_in[0];
  const float* conv1_w = (const float*)d_in[1];
  const float* conv1_b = (const float*)d_in[2];
  const float* bn1_g   = (const float*)d_in[3];
  const float* bn1_b   = (const float*)d_in[4];
  const float* bn1_m   = (const float*)d_in[5];
  const float* bn1_v   = (const float*)d_in[6];
  const float* conv2_w = (const float*)d_in[7];
  const float* conv2_b = (const float*)d_in[8];
  const float* bn2_g   = (const float*)d_in[9];
  const float* bn2_b   = (const float*)d_in[10];
  const float* bn2_m   = (const float*)d_in[11];
  const float* bn2_v   = (const float*)d_in[12];
  const float* fwih = (const float*)d_in[13];
  const float* fbih = (const float*)d_in[14];
  const float* fwhh = (const float*)d_in[15];
  const float* fbhh = (const float*)d_in[16];
  const float* flnih_g = (const float*)d_in[17];
  const float* flnih_b = (const float*)d_in[18];
  const float* flnhh_g = (const float*)d_in[19];
  const float* flnhh_b = (const float*)d_in[20];
  const float* flnho_g = (const float*)d_in[21];
  const float* flnho_b = (const float*)d_in[22];
  const float* bwih = (const float*)d_in[23];
  const float* bbih = (const float*)d_in[24];
  const float* bwhh = (const float*)d_in[25];
  const float* bbhh = (const float*)d_in[26];
  const float* blnih_g = (const float*)d_in[27];
  const float* blnih_b = (const float*)d_in[28];
  const float* blnhh_g = (const float*)d_in[29];
  const float* blnhh_b = (const float*)d_in[30];
  const float* blnho_g = (const float*)d_in[31];
  const float* blnho_b = (const float*)d_in[32];
  const float* attn_w = (const float*)d_in[33];
  const float* fc1_w  = (const float*)d_in[34];
  const float* fc1_b  = (const float*)d_in[35];
  const float* bn3_g  = (const float*)d_in[36];
  const float* bn3_b  = (const float*)d_in[37];
  const float* bn3_m  = (const float*)d_in[38];
  const float* bn3_v  = (const float*)d_in[39];
  const float* fc2_w  = (const float*)d_in[40];
  const float* fc2_b  = (const float*)d_in[41];
  float* out = (float*)d_out;

  char* ws = (char*)d_ws;
  __bf16* act1 = (__bf16*)(ws + OFF_ACT1);
  __bf16* act2 = (__bf16*)(ws + OFF_ACT2);
  float*  xp   = (float*)(ws + OFF_XP);
  float*  lo   = (float*)(ws + OFF_LO);
  __bf16* w2A  = (__bf16*)(ws + OFF_W2A);
  __bf16* wiA  = (__bf16*)(ws + OFF_WIA);
  __bf16* whA  = (__bf16*)(ws + OFF_WHA);

  // pack weights into WMMA A-fragment layouts (bf16)
  pack_conv2A<<<(N_C2A + 255) / 256, 256, 0, stream>>>(conv2_w, w2A);
  pack_wihA<<<(N_WIA + 255) / 256, 256, 0, stream>>>(fwih, wiA);
  pack_wihA<<<(N_WIA + 255) / 256, 256, 0, stream>>>(bwih, wiA + N_WIA);
  pack_whhA<<<(N_WHA + 255) / 256, 256, 0, stream>>>(fwhh, whA);
  pack_whhA<<<(N_WHA + 255) / 256, 256, 0, stream>>>(bwhh, whA + N_WHA);

  // conv1 + bn + relu + pool -> act1 (channel-last bf16)
  conv1_bn_pool<<<(BATCH * T1 * CF1) / 256, 256, 0, stream>>>(
      x, conv1_w, conv1_b, bn1_g, bn1_b, bn1_m, bn1_v, act1);

  // conv2 implicit GEMM: TDM strip tile + WMMA + fused BN/ReLU/pool -> act2
  conv2_wmma_tdm<<<BATCH * 8, 256, 0, stream>>>(
      act1, w2A, conv2_b, bn2_g, bn2_b, bn2_m, bn2_v, act2);

  // xp = LN(seq @ Wih^T + bih) for both directions
  xp_gemm<<<(2 * T2 * 24 * 8) / 8, 256, 0, stream>>>(act2, wiA, fbih, bbih, xp);
  xp_ln<<<(2 * T2 * BATCH * 32) / 256, 256, 0, stream>>>(xp, flnih_g, flnih_b, blnih_g, blnih_b);

  // persistent bidirectional layernorm-LSTM scan (LDS-resident state, WMMA recurrence)
  size_t scan_lds = (size_t)BATCH * G4 * 4 + (size_t)BATCH * HID * 4 + (size_t)BATCH * HID * 2;
  lstm_scan<<<2, 1024, scan_lds, stream>>>(
      xp, whA,
      fbhh, flnhh_g, flnhh_b, flnho_g, flnho_b,
      bbhh, blnhh_g, blnhh_b, blnho_g, blnho_b,
      lo);

  // attention + fc head
  head<<<BATCH, 256, 0, stream>>>(lo, attn_w, fc1_w, fc1_b,
                                  bn3_g, bn3_b, bn3_m, bn3_v, fc2_w, fc2_b, out);
}
